// SelfAttention_32341103739204
// MI455X (gfx1250) — matrix operand order repaired and checked by reading, not compiled, not run on hardware
//
#include <hip/hip_runtime.h>
#include <hip/hip_bf16.h>

// ---------------------------------------------------------------------------
// SAGAN self-attention for MI455X (gfx1250, wave32, WMMA bf16 16x16x32).
// B=4, C=256, N=4096, CK=32.
//
//  prep:   Wt  = [Wf|Wg|Wh]^T as bf16 [320][C];  xT = x^T as bf16 [B][N][C]
//  proj:   fT,gT (bf16 [B][N][32]), hB (bf16 [B][C][N]) via WMMA, all operand
//          fragments are two b128 loads (K-contiguous layouts).
//  stats:  s = g^T f row max / sum, f tiles staged to LDS once per 128 cols
//          with GLOBAL_LOAD_ASYNC_TO_LDS_B128 (8x dedup across waves).
//  out:    recompute p tiles (g tile async-staged to LDS, p packed to LDS in
//          B-fragment uint32 layout), o = h @ p, y = gamma*o + x.
//
// attn [B,N,N] (268 MB) never touches HBM; ~19 MB of intermediates are
// L2-resident (192 MB L2);  HBM traffic ~ x + y only -> bf16-WMMA bound.
// ---------------------------------------------------------------------------

typedef __attribute__((ext_vector_type(16))) __bf16 v16bf;
typedef __attribute__((ext_vector_type(8)))  float  v8f;
typedef int b128i __attribute__((vector_size(16)));   // matches builtin param

#define BN  4
#define CC  256
#define CKK 32
#define NN  4096

union FragU { v16bf v; uint4 q[2]; unsigned u[8]; };

static __device__ __forceinline__ v8f wmma_bf16(v16bf a, v16bf b, v8f c) {
  return __builtin_amdgcn_wmma_f32_16x16x32_bf16(false, a, false, b, (short)0, c,
                                                 false, false);
}

// A-frag (16x32 bf16): lane(half,lm) gets K = [16h, 16h+16) from row lm (this code's k order, same as B)
// -> two 16B chunks at byte offsets 32h and 32h+16 of a 64B row.
static __device__ __forceinline__ v16bf load_a_frag(const __bf16* row, int half) {
  FragU u;
  u.q[0] = *(const uint4*)(row + 16 * half);
  u.q[1] = *(const uint4*)(row + 16 * half + 8);
  return u.v;
}
// B-frag (32x16 bf16): lane(half,lm) gets K = [16h, 16h+16) of column lm (same k order as A)
// -> one contiguous 32B chunk (two uint4) when K is the contiguous axis.
static __device__ __forceinline__ v16bf load_b_frag(const __bf16* col, int half) {
  FragU u;
  u.q[0] = *(const uint4*)(col + 16 * half);
  u.q[1] = *(const uint4*)(col + 16 * half + 8);
  return u.v;
}

static __device__ __forceinline__ unsigned pack2(float a, float b) {
  union { __bf16 h[2]; unsigned u; } t;
  t.h[0] = (__bf16)a; t.h[1] = (__bf16)b;
  return t.u;
}

// ---- CDNA5 async global->LDS copy (ASYNCcnt path) --------------------------
#if __has_builtin(__builtin_amdgcn_global_load_async_to_lds_b128)
#define ASYNC_BUILTIN 1
#else
#define ASYNC_BUILTIN 0
#endif

static __device__ __forceinline__ void async_copy_b128(const void* gsrc, void* ldst) {
#if ASYNC_BUILTIN
  __builtin_amdgcn_global_load_async_to_lds_b128(
      (__attribute__((address_space(1))) b128i*)gsrc,
      (__attribute__((address_space(3))) b128i*)ldst, 0, 0);
#else
  unsigned loff = (unsigned)(unsigned long long)
      (__attribute__((address_space(3))) void*)ldst;
  asm volatile("global_load_async_to_lds_b128 %0, %1, off"
               :: "v"(loff), "v"((unsigned long long)gsrc) : "memory");
#endif
}
static __device__ __forceinline__ void wait_async0() {
#if __has_builtin(__builtin_amdgcn_s_wait_asynccnt)
  __builtin_amdgcn_s_wait_asynccnt(0);
#else
  asm volatile("s_wait_asynccnt 0" ::: "memory");
#endif
}

// ---------------------------------------------------------------------------
// prep 1: Wt[j][c] = bf16(W[c][j]) for j in [0,320) = [Wf|Wg|Wh].
// ---------------------------------------------------------------------------
__global__ __launch_bounds__(256) void prep_w_kernel(
    const float* __restrict__ Wf, const float* __restrict__ Wg,
    const float* __restrict__ Wh, __bf16* __restrict__ Wt)
{
  int j = blockIdx.x, c = threadIdx.x;
  float v;
  if (j < 32)      v = Wf[c * CKK + j];
  else if (j < 64) v = Wg[c * CKK + (j - 32)];
  else             v = Wh[c * CC + (j - 64)];
  Wt[j * CC + c] = (__bf16)v;
}

// ---------------------------------------------------------------------------
// prep 2: xT[b][n][c] = bf16(x[b][c][n]).  32x32 LDS transpose tiles.
// ---------------------------------------------------------------------------
__global__ __launch_bounds__(256) void transpose_x_kernel(
    const float* __restrict__ x, __bf16* __restrict__ xT)
{
  __shared__ float t[32][33];
  const int b = blockIdx.z, c0 = blockIdx.y * 32, n0 = blockIdx.x * 32;
  const int tid = threadIdx.x;
  #pragma unroll
  for (int p = 0; p < 4; ++p) {
    int i = (tid >> 5) + p * 8, j = tid & 31;
    t[i][j] = x[((size_t)(b * CC + c0 + i)) * NN + n0 + j];
  }
  __syncthreads();
  #pragma unroll
  for (int p = 0; p < 2; ++p) {
    int q = tid + p * 256;            // 512 uint pairs
    int n = q >> 4, cp = q & 15;      // pair of channels
    unsigned u = pack2(t[2 * cp][n], t[2 * cp + 1][n]);
    ((unsigned*)(xT + ((size_t)(b * NN + n0 + n)) * CC + c0))[cp] = u;
  }
}

// ---------------------------------------------------------------------------
// Pass 1: projections. grid (N/128, 320/64, B), 256 thr. No LDS: fragments
// come straight from xT / Wt as b128 loads.
// ---------------------------------------------------------------------------
__global__ __launch_bounds__(256) void proj_kernel(
    const float* __restrict__ bfv, const float* __restrict__ bgv,
    const float* __restrict__ bhv,
    const __bf16* __restrict__ xT, const __bf16* __restrict__ Wt,
    __bf16* __restrict__ fT, __bf16* __restrict__ gT, __bf16* __restrict__ hB)
{
  const int b = blockIdx.z, n0 = blockIdx.x * 128, j0 = blockIdx.y * 64;
  const int tid = threadIdx.x, lane = tid & 31, wave = tid >> 5;
  const int half = lane >> 4, lm = lane & 15;
  const int jt = wave & 3, mtb = (wave >> 2) * 4;
  const int jcol = j0 + jt * 16 + lm;

  v8f acc[4] = {};
  const __bf16* wcol = Wt + (size_t)jcol * CC;

  for (int c0 = 0; c0 < CC; c0 += 32) {
    v16bf bfrag = load_b_frag(wcol + c0, half);
    #pragma unroll
    for (int t = 0; t < 4; ++t) {
      int n = n0 + (mtb + t) * 16 + lm;
      v16bf afrag = load_a_frag(xT + ((size_t)(b * NN + n)) * CC + c0, half);
      acc[t] = wmma_bf16(afrag, bfrag, acc[t]);
    }
  }

  float bias;
  if (jcol < 32)      bias = bfv[jcol];
  else if (jcol < 64) bias = bgv[jcol - 32];
  else                bias = bhv[jcol - 64];

  #pragma unroll
  for (int t = 0; t < 4; ++t) {
    if (jcol < 64) {
      #pragma unroll
      for (int r = 0; r < 8; ++r) {
        int   n = n0 + (mtb + t) * 16 + half * 8 + r;
        float v = acc[t][r] + bias;
        v = v > 0.f ? v : 0.f;
        if (jcol < 32) fT[((size_t)b * NN + n) * CKK + jcol] = (__bf16)v;
        else           gT[((size_t)b * NN + n) * CKK + (jcol - 32)] = (__bf16)v;
      }
    } else {                       // h rows: n-consecutive -> pack pairs
      #pragma unroll
      for (int rp = 0; rp < 4; ++rp) {
        int   n  = n0 + (mtb + t) * 16 + half * 8 + 2 * rp;
        float v0 = acc[t][2 * rp]     + bias; v0 = v0 > 0.f ? v0 : 0.f;
        float v1 = acc[t][2 * rp + 1] + bias; v1 = v1 > 0.f ? v1 : 0.f;
        *(unsigned*)(hB + ((size_t)(b * CC + (jcol - 64)) * NN) + n) = pack2(v0, v1);
      }
    }
  }
}

// ---------------------------------------------------------------------------
// Pass 2: row stats of s = g^T f. grid (N/128, B), 256 thr.
// f columns staged to LDS in 128-col chunks via async copy (8x wave dedup).
// ---------------------------------------------------------------------------
__global__ __launch_bounds__(256) void stats_kernel(
    const __bf16* __restrict__ fT, const __bf16* __restrict__ gT,
    float* __restrict__ Mrow, float* __restrict__ Zinv)
{
  __shared__ __align__(16) char fS[128 * 80];   // 128 rows x 32 bf16, 80B pitch
  const int b = blockIdx.y;
  const int tid = threadIdx.x, lane = tid & 31, wave = tid >> 5;
  const int half = lane >> 4, lm = lane & 15;
  const int n0 = blockIdx.x * 128 + wave * 16;

  v16bf a = load_a_frag(gT + ((size_t)b * NN + (n0 + lm)) * CKK, half);

  float rmax[8], rsum[8];
  #pragma unroll
  for (int r = 0; r < 8; ++r) { rmax[r] = -1e30f; rsum[r] = 0.f; }

  for (int mb = 0; mb < NN; mb += 128) {
    // async-stage fT[b, mb..mb+127, :] (contiguous 8 KB) into padded LDS
    const char* src = (const char*)(fT + ((size_t)b * NN + mb) * CKK);
    #pragma unroll
    for (int i = 0; i < 2; ++i) {
      int seg = tid + i * 256;                        // 16B segments
      async_copy_b128(src + seg * 16, fS + (seg >> 2) * 80 + (seg & 3) * 16);
    }
    wait_async0();
    __syncthreads();

    #pragma unroll
    for (int mi = 0; mi < 8; ++mi) {
      v16bf bb = load_b_frag((const __bf16*)(fS + (mi * 16 + lm) * 80), half);
      v8f zero = {};
      v8f s = wmma_bf16(a, bb, zero);
      #pragma unroll
      for (int r = 0; r < 8; ++r) {
        float v  = s[r];
        float nm = fmaxf(rmax[r], v);
        rsum[r]  = rsum[r] * __expf(rmax[r] - nm) + __expf(v - nm);
        rmax[r]  = nm;
      }
    }
    __syncthreads();
  }

  #pragma unroll
  for (int r = 0; r < 8; ++r) {
    #pragma unroll
    for (int mask = 1; mask < 16; mask <<= 1) {
      float om = __shfl_xor(rmax[r], mask, 32);
      float os = __shfl_xor(rsum[r], mask, 32);
      float nm = fmaxf(rmax[r], om);
      rsum[r]  = rsum[r] * __expf(rmax[r] - nm) + os * __expf(om - nm);
      rmax[r]  = nm;
    }
    if (lm == 0) {
      int n = n0 + half * 8 + r;
      Mrow[(size_t)b * NN + n] = rmax[r];
      Zinv[(size_t)b * NN + n] = 1.f / rsum[r];
    }
  }
}

// ---------------------------------------------------------------------------
// Pass 3: o = h @ p, y = gamma*o + x. grid (N/128, C/64, B), 256 thr.
// ---------------------------------------------------------------------------
__global__ __launch_bounds__(256) void attn_out_kernel(
    const float* __restrict__ x, const float* __restrict__ gammap,
    const __bf16* __restrict__ fT, const __bf16* __restrict__ gT,
    const __bf16* __restrict__ hB,
    const float* __restrict__ Mrow, const float* __restrict__ Zinv,
    float* __restrict__ y)
{
  __shared__ __align__(16) char gS[32 * 80];   // g tile: 32 rows, 80B pitch
  __shared__ unsigned pS[16 * 132];            // p tile packed pairs [pairRow][m]
  const int b = blockIdx.z, m0 = blockIdx.x * 128, c0 = blockIdx.y * 64;
  const int tid = threadIdx.x, lane = tid & 31, wave = tid >> 5;
  const int half = lane >> 4, lm = lane & 15;
  const int ct = wave & 3, mtb = (wave >> 2) * 4;

  v8f acc[4] = {};
  const float gamma = gammap[0];

  // f B-fragment for this wave's p-slice columns (constant over n loop)
  const int mc = m0 + wave * 16 + lm;
  v16bf bf_f = load_b_frag(fT + ((size_t)b * NN + mc) * CKK, half);

  const __bf16* hrow = hB + ((size_t)(b * CC + (c0 + ct * 16 + lm)) * NN);

  for (int nb = 0; nb < NN; nb += 32) {
    // -- async-stage gT rows nb..nb+31 (contiguous 2 KB) into padded LDS
    if (wave == 0) {
      const char* src = (const char*)(gT + ((size_t)b * NN + nb) * CKK);
      #pragma unroll
      for (int i = 0; i < 4; ++i) {
        int seg = lane + i * 32;
        async_copy_b128(src + seg * 16, gS + (seg >> 2) * 80 + (seg & 3) * 16);
      }
      wait_async0();
    }
    __syncthreads();

    // -- stage 1: p[nb..nb+31][m0..m0+127] -> pS (packed B-frag layout)
    #pragma unroll
    for (int nh = 0; nh < 2; ++nh) {
      int nbase = nb + nh * 16;
      v16bf ag = load_a_frag((const __bf16*)(gS + (nh * 16 + lm) * 80), half);
      v8f zero = {};
      v8f s = wmma_bf16(ag, bf_f, zero);

      float mv[8], zv[8];
      *(float4*)&mv[0] = *(const float4*)(Mrow + (size_t)b * NN + nbase + 8 * half);
      *(float4*)&mv[4] = *(const float4*)(Mrow + (size_t)b * NN + nbase + 8 * half + 4);
      *(float4*)&zv[0] = *(const float4*)(Zinv + (size_t)b * NN + nbase + 8 * half);
      *(float4*)&zv[4] = *(const float4*)(Zinv + (size_t)b * NN + nbase + 8 * half + 4);

      #pragma unroll
      for (int rp = 0; rp < 4; ++rp) {
        float p0 = __expf(s[2 * rp]     - mv[2 * rp])     * zv[2 * rp];
        float p1 = __expf(s[2 * rp + 1] - mv[2 * rp + 1]) * zv[2 * rp + 1];
        pS[(nh * 8 + half * 4 + rp) * 132 + (wave * 16 + lm)] = pack2(p0, p1);
      }
    }
    __syncthreads();

    // -- stage 2: o tiles += h(16c x 32n) * p(32n x 16m)
    v16bf ah = load_a_frag(hrow + nb, half);
    if (nb + 32 < NN) __builtin_prefetch((const void*)(hrow + nb + 32), 0, 0);
    #pragma unroll
    for (int t = 0; t < 4; ++t) {
      int mcol = (mtb + t) * 16 + lm;
      FragU u;
      #pragma unroll
      for (int j = 0; j < 8; ++j)
        u.u[j] = pS[(half * 8 + j) * 132 + mcol];
      acc[t] = wmma_bf16(ah, u.v, acc[t]);
    }
    __syncthreads();
  }

  #pragma unroll
  for (int t = 0; t < 4; ++t) {
    #pragma unroll
    for (int r = 0; r < 8; ++r) {
      int c = c0 + ct * 16 + half * 8 + r;
      int m = m0 + (mtb + t) * 16 + lm;
      size_t idx = ((size_t)(b * CC + c)) * NN + m;
      y[idx] = gamma * acc[t][r] + x[idx];
    }
  }
}

// ---------------------------------------------------------------------------
// Workspace (bytes): fT@0(1M)  gT@1M(1M)  hB@2M(8M)  xT@10M(8M)
//                    Wt@18M(160K)  Mrow@18M+256K(64K)  Zinv@+64K
// ---------------------------------------------------------------------------
extern "C" void kernel_launch(void* const* d_in, const int* in_sizes, int n_in,
                              void* d_out, int out_size, void* d_ws, size_t ws_size,
                              hipStream_t stream) {
  const float* x     = (const float*)d_in[0];
  const float* Wf    = (const float*)d_in[1];
  const float* bfv   = (const float*)d_in[2];
  const float* Wg    = (const float*)d_in[3];
  const float* bgv   = (const float*)d_in[4];
  const float* Wh    = (const float*)d_in[5];
  const float* bhv   = (const float*)d_in[6];
  const float* gamma = (const float*)d_in[7];
  float* y = (float*)d_out;

  char* ws = (char*)d_ws;
  __bf16* fT   = (__bf16*)(ws);
  __bf16* gT   = (__bf16*)(ws + ((size_t)1 << 20));
  __bf16* hB   = (__bf16*)(ws + ((size_t)2 << 20));
  __bf16* xT   = (__bf16*)(ws + ((size_t)10 << 20));
  __bf16* Wt   = (__bf16*)(ws + ((size_t)18 << 20));
  float*  Mrow = (float*) (ws + ((size_t)18 << 20) + 0x40000);
  float*  Zinv = (float*) (ws + ((size_t)18 << 20) + 0x50000);

  prep_w_kernel<<<dim3(320), 256, 0, stream>>>(Wf, Wg, Wh, Wt);
  transpose_x_kernel<<<dim3(NN / 32, CC / 32, BN), 256, 0, stream>>>(x, xT);
  proj_kernel<<<dim3(NN / 128, 320 / 64, BN), 256, 0, stream>>>(
      bfv, bgv, bhv, xT, Wt, fT, gT, hB);
  stats_kernel<<<dim3(NN / 128, BN), 256, 0, stream>>>(fT, gT, Mrow, Zinv);
  attn_out_kernel<<<dim3(NN / 128, CC / 64, BN), 256, 0, stream>>>(
      x, gamma, fT, gT, hB, Mrow, Zinv, y);
}
